// HungarianMatcher_89610197664154
// MI455X (gfx1250) — matrix operand order repaired and checked by reading, not compiled, not run on hardware
//
#include <hip/hip_runtime.h>

// ---------------- problem constants ----------------
#define B_    32
#define Q_    900
#define C_    92
#define T_    2048
#define QT_   57            // ceil(900/16) query tiles per batch
#define KPAD_ 128           // 92 padded to 4 x 32 for bf16 WMMA
#define NTT_  128           // 2048/16 target tiles

typedef __attribute__((ext_vector_type(16))) __bf16 v16bf;
typedef __attribute__((ext_vector_type(8)))  float  v8f;

// round-to-nearest-even float -> bf16 bits
__device__ __forceinline__ unsigned short f2bf(float f) {
  unsigned int x = __float_as_uint(f);
  x += 0x7fffu + ((x >> 16) & 1u);
  return (unsigned short)(x >> 16);
}

// ---------------------------------------------------------------------------
// Kernel 1: softmax probabilities, written pre-swizzled into the WMMA A-matrix
// register layout (16-bit A 16x32, ISA 7.12.2):
//   lanes 0-15 : M=lane,    elems i=0..7 -> K=i,     i=8..15 -> K=i+8
//   lanes 16-31: M=lane-16, elems i=0..7 -> K=i+8,   i=8..15 -> K=i+16
// ws chunk = 16 bf16 (32B) per (b,qtile,kb,lane); zeros for K>=92 or q>=900.
// One thread per (b, qtile, m).
// ---------------------------------------------------------------------------
__global__ void prep_probs(const float* __restrict__ logits,
                           unsigned short* __restrict__ wsA) {
  int tid = blockIdx.x * blockDim.x + threadIdx.x;
  if (tid >= B_ * QT_ * 16) return;
  int m  = tid & 15;
  int qt = (tid >> 4) % QT_;
  int b  = tid / (QT_ * 16);
  int q  = qt * 16 + m;
  bool valid = (q < Q_);
  const float* row = logits + (size_t)(b * Q_ + (valid ? q : 0)) * C_;

  float mx = -3.0e38f;
  for (int c = 0; c < C_; ++c) mx = fmaxf(mx, row[c]);
  float s = 0.0f;
  for (int c = 0; c < C_; ++c) s += __expf(row[c] - mx);
  float rs = __builtin_amdgcn_rcpf(s);

  for (int c = 0; c < KPAD_; c += 2) {
    float p0 = (valid && (c + 0) < C_) ? __expf(row[c + 0] - mx) * rs : 0.0f;
    float p1 = (valid && (c + 1) < C_) ? __expf(row[c + 1] - mx) * rs : 0.0f;
    unsigned int packed = (unsigned int)f2bf(p0) | ((unsigned int)f2bf(p1) << 16);
    int kb = c >> 5, kk = c & 31;
    int lane_hi = (kk >> 3) & 1;                                  // which half-wave holds this K
    int i = (kk < 8) ? kk : (kk < 24 ? kk - 8 : kk - 16);         // element index within lane
    int lane = lane_hi * 16 + m;
    size_t chunk = ((size_t)((b * QT_ + qt) * 4 + kb)) * 32 + lane;
    *(unsigned int*)(wsA + chunk * 16 + i) = packed;
  }
}

// ---------------------------------------------------------------------------
// Kernel 2: one-hot(label) B tiles in the WMMA B-matrix register layout
// (16-bit B 32x16, same K order as A: lanes 0-15 N=lane K=0..7,16..23; lanes 16-31 N=lane-16 K=8..15,24..31).
// Shared across all batches/queries: 128 ttiles x 4 kb. One thread per
// (ttile, kb, lane); each thread emits its 16 bf16 (32B) chunk.
// ---------------------------------------------------------------------------
__global__ void prep_onehot(const int* __restrict__ labels,
                            unsigned short* __restrict__ wsB) {
  int tid = blockIdx.x * blockDim.x + threadIdx.x;
  if (tid >= NTT_ * 4 * 32) return;
  int lane = tid & 31;
  int kb   = (tid >> 5) & 3;
  int tt   = tid >> 7;
  int t     = tt * 16 + (lane & 15);
  int label = labels[t];
  int kbase = kb * 32 + ((lane < 16) ? 0 : 8);
  unsigned int w[8];
#pragma unroll
  for (int j = 0; j < 8; ++j) {
    unsigned int lo = (label == kbase + (j < 4 ? 2 * j : 8 + 2 * j) + 0) ? 0x3F80u : 0u;  // bf16(1.0)
    unsigned int hi = (label == kbase + (j < 4 ? 2 * j : 8 + 2 * j) + 1) ? 0x3F80u : 0u;
    w[j] = lo | (hi << 16);
  }
  uint4* dst = (uint4*)(wsB + ((size_t)(tt * 4 + kb) * 32 + lane) * 16);
  dst[0] = make_uint4(w[0], w[1], w[2], w[3]);
  dst[1] = make_uint4(w[4], w[5], w[6], w[7]);
}

// ---------------------------------------------------------------------------
// per-element box cost: 5*L1 - 2*GIoU - prob  (v_rcp_f32 for the divides)
// ---------------------------------------------------------------------------
__device__ __forceinline__ float elem_cost(float4 pb, float4 tb,
                                           float tx0, float ty0,
                                           float tx1, float ty1,
                                           float area2, float cls) {
  float phw = 0.5f * pb.z, phh = 0.5f * pb.w;
  float px0 = pb.x - phw, py0 = pb.y - phh;
  float px1 = pb.x + phw, py1 = pb.y + phh;
  float area1 = pb.z * pb.w;

  float l1 = fabsf(pb.x - tb.x) + fabsf(pb.y - tb.y) +
             fabsf(pb.z - tb.z) + fabsf(pb.w - tb.w);

  float ix0 = fmaxf(px0, tx0), iy0 = fmaxf(py0, ty0);
  float ix1 = fminf(px1, tx1), iy1 = fminf(py1, ty1);
  float iw = fmaxf(ix1 - ix0, 0.0f), ih = fmaxf(iy1 - iy0, 0.0f);
  float inter = iw * ih;
  float uni = area1 + area2 - inter;
  float iou = inter * __builtin_amdgcn_rcpf(uni);

  float ex0 = fminf(px0, tx0), ey0 = fminf(py0, ty0);
  float ex1 = fmaxf(px1, tx1), ey1 = fmaxf(py1, ty1);
  float ew = fmaxf(ex1 - ex0, 0.0f), eh = fmaxf(ey1 - ey0, 0.0f);
  float areaE = ew * eh;
  float giou = iou - (areaE - uni) * __builtin_amdgcn_rcpf(areaE);

  return 5.0f * l1 - 2.0f * giou - cls;   // 1*(-prob) + 5*l1 + 2*(-giou)
}

// ---------------------------------------------------------------------------
// Kernel 3: main cost kernel. One wave per 16(Q) x 16(T) tile.
// grid = (NTT_/8, QT_, B_): no integer div/mod, boundary check is
// block-uniform (only qt == QT_-1 has padded rows).
//   class cost: 4 x v_wmma_f32_16x16x32_bf16 (P @ OneHot^T), K = 128
//   bbox/giou : VALU, one target column per lane
// D layout (ISA): lane holds N=lane%16, rows M = r + (lane<16 ? 0 : 8).
// Output is write-once -> nontemporal stores keep L2 for the reused tiles.
// ---------------------------------------------------------------------------
__global__ void cost_main(const float4* __restrict__ pred_boxes,   // [B*Q]
                          const float4* __restrict__ tgt_boxes,    // [T]
                          const unsigned short* __restrict__ wsA,
                          const unsigned short* __restrict__ wsB,
                          float* __restrict__ out) {
  const int lane = threadIdx.x & 31;
  const int tt = blockIdx.x * 8 + (threadIdx.x >> 5);
  const int qt = blockIdx.y;
  const int b  = blockIdx.z;

  const v16bf* Ap = (const v16bf*)wsA + (size_t)((b * QT_ + qt) * 4) * 32 + lane;
  const v16bf* Bp = (const v16bf*)wsB + (size_t)(tt * 4) * 32 + lane;

  v8f acc = {};
#pragma unroll
  for (int kb = 0; kb < 4; ++kb) {
    v16bf a  = Ap[kb * 32];
    v16bf bm = Bp[kb * 32];
    acc = __builtin_amdgcn_wmma_f32_16x16x32_bf16(
        /*neg_a=*/false, a, /*neg_b=*/false, bm,
        /*c_mod=*/(short)0, acc, /*reuse_a=*/false, /*reuse_b=*/false);
  }

  // per-lane target (fixed column), hoisted out of the row loop
  const int t = tt * 16 + (lane & 15);
  float4 tb = tgt_boxes[t];
  float thw = 0.5f * tb.z, thh = 0.5f * tb.w;
  float tx0 = tb.x - thw, ty0 = tb.y - thh;
  float tx1 = tb.x + thw, ty1 = tb.y + thh;
  float area2 = tb.z * tb.w;

  const int mbase = (lane < 16) ? 0 : 8;
  const int qbase = qt * 16 + mbase;
  const float4* pbp = pred_boxes + (size_t)b * Q_ + qbase;
  float* op = out + (size_t)(b * Q_ + qbase) * T_ + t;

  if (qt != QT_ - 1) {
    // fast path: all 16 rows valid; affine addresses fold into imm offsets
#pragma unroll
    for (int r = 0; r < 8; ++r) {
      float c = elem_cost(pbp[r], tb, tx0, ty0, tx1, ty1, area2, acc[r]);
      __builtin_nontemporal_store(c, op + (size_t)r * T_);
    }
  } else {
    // tail tile: rows q >= 900 are padding (A rows are zero there)
#pragma unroll
    for (int r = 0; r < 8; ++r) {
      int q = qbase + r;
      int ridx = (q < Q_) ? r : (Q_ - 1 - qbase);   // clamp load to a valid box
      float c = elem_cost(pbp[ridx], tb, tx0, ty0, tx1, ty1, area2, acc[r]);
      if (q < Q_)
        __builtin_nontemporal_store(c, op + (size_t)r * T_);
    }
  }
}

// ---------------------------------------------------------------------------
extern "C" void kernel_launch(void* const* d_in, const int* in_sizes, int n_in,
                              void* d_out, int out_size, void* d_ws, size_t ws_size,
                              hipStream_t stream) {
  (void)in_sizes; (void)n_in; (void)out_size; (void)ws_size;
  const float*  logits = (const float*)d_in[0];
  const float4* pboxes = (const float4*)d_in[1];
  const int*    labels = (const int*)d_in[2];
  const float4* tboxes = (const float4*)d_in[3];
  float* out = (float*)d_out;

  // ws: A (swizzled bf16 probs) 7.47 MB, then B (one-hot tiles) 0.52 MB
  unsigned short* wsA = (unsigned short*)d_ws;
  unsigned short* wsB = wsA + (size_t)B_ * QT_ * 4 * 32 * 16;

  int nA = B_ * QT_ * 16;                 // 29184 threads
  prep_probs<<<(nA + 255) / 256, 256, 0, stream>>>(logits, wsA);

  int nB = NTT_ * 4 * 32;                 // 16384 threads
  prep_onehot<<<(nB + 255) / 256, 256, 0, stream>>>(labels, wsB);

  // one wave per 16x16 tile: 8 waves/block over tt, grid (16, 57, 32)
  dim3 grid(NTT_ / 8, QT_, B_);
  cost_main<<<grid, 256, 0, stream>>>(pboxes, tboxes, wsA, wsB, out);
}
